// RN5_54820962566530
// MI455X (gfx1250) — matrix operand order repaired and checked by reading, not compiled, not run on hardware
//
#include <hip/hip_runtime.h>

#define V_DIMC 2048
#define Q_DIMC 1024
#define N_OBJC 36
#define BATCHC 64

typedef __attribute__((ext_vector_type(16))) __bf16 v16bf;
typedef __attribute__((ext_vector_type(8)))  __bf16 v8bf;
typedef __attribute__((ext_vector_type(8)))  float  v8f;
typedef __attribute__((ext_vector_type(4)))  float  v4f;
typedef __attribute__((ext_vector_type(4)))  int    v4i;

// pointer-to-addrspace typedefs (attribute applies to the pointee)
typedef __attribute__((address_space(1))) v4i* as1_v4i_p;
typedef __attribute__((address_space(3))) v4i* as3_v4i_p;

#if __has_builtin(__builtin_amdgcn_global_load_async_to_lds_b128) && \
    __has_builtin(__builtin_amdgcn_s_wait_asynccnt)
#define USE_ASYNC_LDS 1
#else
#define USE_ASYNC_LDS 0
#endif

__device__ __forceinline__ __bf16 f2bf(float f) {
  // round-to-nearest-even truncation to bf16 (integer path, always lowers)
  unsigned u = __builtin_bit_cast(unsigned, f);
  u += 0x7FFFu + ((u >> 16) & 1u);
  unsigned short h = (unsigned short)(u >> 16);
  return __builtin_bit_cast(__bf16, h);
}

// 16-byte global->LDS copy: async (ASYNCcnt, no VGPR round trip) when available.
__device__ __forceinline__ void cp16(const __bf16* g, __bf16* l) {
#if USE_ASYNC_LDS
  __builtin_amdgcn_global_load_async_to_lds_b128(
      (as1_v4i_p)(v4i*)g, (as3_v4i_p)(v4i*)l, 0, 0);
#else
  *reinterpret_cast<v8bf*>(l) = *reinterpret_cast<const v8bf*>(g);
#endif
}

__device__ __forceinline__ void wait_async_all() {
#if USE_ASYNC_LDS
  __builtin_amdgcn_s_wait_asynccnt(0);
#endif
}

// ---------------------------------------------------------------------------
// one-time f32 -> bf16 pre-convert (8 elems / thread)
// ---------------------------------------------------------------------------
__global__ __launch_bounds__(256)
void cvt_f32_bf16(const float* __restrict__ src, __bf16* __restrict__ dst) {
  const size_t i = ((size_t)blockIdx.x * 256 + threadIdx.x) * 8;
  v4f a = *reinterpret_cast<const v4f*>(src + i);
  v4f b = *reinterpret_cast<const v4f*>(src + i + 4);
  v8bf o;
  o[0] = f2bf(a[0]); o[1] = f2bf(a[1]); o[2] = f2bf(a[2]); o[3] = f2bf(a[3]);
  o[4] = f2bf(b[0]); o[5] = f2bf(b[1]); o[6] = f2bf(b[2]); o[7] = f2bf(b[3]);
  *reinterpret_cast<v8bf*>(dst + i) = o;
}

// ---------------------------------------------------------------------------
// C[m,n] = sum_k A[m,k]*B[n,k]  (+bias[n]) (optional relu).  A,B bf16 row-major.
// 128 threads = 4 waves; block tile 64x64; K-step 32; double-buffered LDS
// filled by GLOBAL_LOAD_ASYNC_TO_LDS_B128 (fallback: vector copy).
// LDS rows padded to 40 halves (80B) for bank spread; 16B alignment preserved.
// ---------------------------------------------------------------------------
#define LDSROW 40

__global__ __launch_bounds__(128)
void gemm_nt_wmma(const __bf16* __restrict__ A, int lda,
                  const __bf16* __restrict__ B, int ldb,
                  const float* __restrict__ bias,
                  float* __restrict__ C, int ldc,
                  int K, int hasBias, int doRelu)
{
  __shared__ __bf16 Asm[2][64 * LDSROW];
  __shared__ __bf16 Bsm[2][64 * LDSROW];

  const int tid  = threadIdx.x;
  const int lane = tid & 31;
  const int wave = tid >> 5;
  const int mBase = blockIdx.y * 64;
  const int nBase = blockIdx.x * 64;

  // fragment lane decode (ISA 7.12.2, wave32)
  const int m  = lane & 15;   // M row (A) / N col (B, C/D) within 16
  const int hp = lane >> 4;   // half-wave selector

  // staging: tile = 64 rows x 32 halves = 256 chunks of 8 halves (16B);
  // thread t copies chunks t and t+128.
  const int r0 = tid >> 2;          // rows 0..31
  const int s0 = (tid & 3) * 8;     // k-segment (halves)
  const int r1 = r0 + 32;           // rows 32..63

  auto stage = [&](int buf, int k0) {
    const __bf16* ga = A + (size_t)mBase * lda + k0;
    cp16(ga + (size_t)r0 * lda + s0, &Asm[buf][r0 * LDSROW + s0]);
    cp16(ga + (size_t)r1 * lda + s0, &Asm[buf][r1 * LDSROW + s0]);
    const __bf16* gb = B + (size_t)nBase * ldb + k0;
    cp16(gb + (size_t)r0 * ldb + s0, &Bsm[buf][r0 * LDSROW + s0]);
    cp16(gb + (size_t)r1 * ldb + s0, &Bsm[buf][r1 * LDSROW + s0]);
  };

  v8f acc[4] = {};

  stage(0, 0);
  for (int k0 = 0; k0 < K; k0 += 32) {
    const int buf = (k0 >> 5) & 1;
    wait_async_all();     // this wave's async copies into Asm/Bsm[buf] done
    __syncthreads();      // everyone's copies done (barrier waits DScnt too)
    if (k0 + 32 < K) stage(buf ^ 1, k0 + 32);   // prefetch next tile

    // A fragment: lanes 0-15 hold K[0..7],K[16..23]; lanes 16-31 K[8..15],K[24..31]
    v16bf afrag;
    {
      const __bf16* base = &Asm[buf][(wave * 16 + m) * LDSROW + hp * 8];
      v8bf a0 = *reinterpret_cast<const v8bf*>(base);        // K[hp*8 .. +7]
      v8bf a1 = *reinterpret_cast<const v8bf*>(base + 16);   // K[16+hp*8 .. +7]
      #pragma unroll
      for (int e = 0; e < 8; ++e) { afrag[e] = a0[e]; afrag[8 + e] = a1[e]; }
    }

    #pragma unroll
    for (int t = 0; t < 4; ++t) {
      // B fragment: lanes 0-15 hold K[0..7],K[16..23]; lanes 16-31 K[8..15],K[24..31]
      const __bf16* base = &Bsm[buf][(t * 16 + m) * LDSROW + hp * 8];
      v8bf b0 = *reinterpret_cast<const v8bf*>(base);
      v8bf b1 = *reinterpret_cast<const v8bf*>(base + 16);
      v16bf bfrag;
      #pragma unroll
      for (int e = 0; e < 8; ++e) { bfrag[e] = b0[e]; bfrag[8 + e] = b1[e]; }

      acc[t] = __builtin_amdgcn_wmma_f32_16x16x32_bf16(
          false, afrag, false, bfrag, (short)0, acc[t], false, false);
    }
  }

  // epilogue: C/D layout: lanes 0-15 -> M=r, N=lane ; lanes 16-31 -> M=8+r, N=lane-16
  #pragma unroll
  for (int t = 0; t < 4; ++t) {
    const int col = nBase + t * 16 + m;
    const float bv = hasBias ? bias[col] : 0.0f;
    #pragma unroll
    for (int r = 0; r < 8; ++r) {
      const int row = mBase + wave * 16 + hp * 8 + r;
      float val = acc[t][r] + bv;
      if (doRelu) val = fmaxf(val, 0.0f);
      C[(size_t)row * ldc + col] = val;
    }
  }
}

// ---------------------------------------------------------------------------
// x[b,i,o] = sum_j relu( u[b,i,o] + u[b,j,o] + c[b,o] )   (x written in bf16)
// ---------------------------------------------------------------------------
__global__ __launch_bounds__(256)
void pairsum_relu(const float* __restrict__ u, const float* __restrict__ c,
                  __bf16* __restrict__ x)
{
  const int o = blockIdx.x * 256 + threadIdx.x;
  const int b = blockIdx.y;
  const float cv = c[(size_t)b * V_DIMC + o];
  const float* ub = u + (size_t)b * N_OBJC * V_DIMC + o;
  float uj[N_OBJC];
  #pragma unroll
  for (int j = 0; j < N_OBJC; ++j) uj[j] = ub[(size_t)j * V_DIMC];
  __bf16* xb = x + (size_t)b * N_OBJC * V_DIMC + o;
  for (int i = 0; i < N_OBJC; ++i) {
    const float ti = uj[i] + cv;
    float s = 0.0f;
    #pragma unroll
    for (int j = 0; j < N_OBJC; ++j) s += fmaxf(ti + uj[j], 0.0f);
    xb[(size_t)i * V_DIMC] = f2bf(s);
  }
}

extern "C" void kernel_launch(void* const* d_in, const int* in_sizes, int n_in,
                              void* d_out, int out_size, void* d_ws, size_t ws_size,
                              hipStream_t stream) {
  const float* v  = (const float*)d_in[0];  // (64, 36, 2048)
  const float* q  = (const float*)d_in[1];  // (64, 1024)
  const float* W1 = (const float*)d_in[2];  // (2048, 3072)
  const float* b1 = (const float*)d_in[3];
  const float* W2 = (const float*)d_in[4];  // (2048, 2048)
  const float* b2 = (const float*)d_in[5];
  float* out = (float*)d_out;               // (64, 36, 2048)

  const size_t M      = (size_t)BATCHC * N_OBJC;               // 2304
  const size_t nV     = M * V_DIMC;                            // 4,718,592
  const size_t nQ     = (size_t)BATCHC * Q_DIMC;               // 65,536
  const size_t nW1    = (size_t)V_DIMC * (V_DIMC + Q_DIMC);    // 6,291,456
  const size_t nW2    = (size_t)V_DIMC * V_DIMC;               // 4,194,304
  const size_t nC     = (size_t)BATCHC * V_DIMC;               // 131,072

  char* ws = (char*)d_ws;
  float*  u   = (float*)ws;                 ws += nV * sizeof(float);
  float*  c   = (float*)ws;                 ws += nC * sizeof(float);
  __bf16* vb  = (__bf16*)ws;                ws += nV  * sizeof(__bf16);
  __bf16* qb  = (__bf16*)ws;                ws += nQ  * sizeof(__bf16);
  __bf16* W1b = (__bf16*)ws;                ws += nW1 * sizeof(__bf16);
  __bf16* W2b = (__bf16*)ws;                ws += nW2 * sizeof(__bf16);
  __bf16* xb  = (__bf16*)ws;

  // one-time bf16 pre-conversion (all sizes divisible by 2048 elems/block)
  cvt_f32_bf16<<<dim3((unsigned)(nV  / 2048)), dim3(256), 0, stream>>>(v,  vb);
  cvt_f32_bf16<<<dim3((unsigned)(nQ  / 2048)), dim3(256), 0, stream>>>(q,  qb);
  cvt_f32_bf16<<<dim3((unsigned)(nW1 / 2048)), dim3(256), 0, stream>>>(W1, W1b);
  cvt_f32_bf16<<<dim3((unsigned)(nW2 / 2048)), dim3(256), 0, stream>>>(W2, W2b);

  dim3 blk(128);

  // c = q @ Wq^T + b1   (M=64, N=2048, K=1024); Wq = W1[:, 2048:], row stride 3072
  gemm_nt_wmma<<<dim3(V_DIMC / 64, BATCHC / 64), blk, 0, stream>>>(
      qb, Q_DIMC, W1b + V_DIMC, V_DIMC + Q_DIMC, b1, c, V_DIMC, Q_DIMC, 1, 0);

  // u = v @ Wv^T        (M=2304, N=2048, K=2048)
  gemm_nt_wmma<<<dim3(V_DIMC / 64, (unsigned)(M / 64)), blk, 0, stream>>>(
      vb, V_DIMC, W1b, V_DIMC + Q_DIMC, nullptr, u, V_DIMC, V_DIMC, 0, 0);

  // x[b,i,o] = sum_j relu(u_i + u_j + c)   (bf16 out)
  pairsum_relu<<<dim3(V_DIMC / 256, BATCHC), dim3(256), 0, stream>>>(u, c, xb);

  // out = relu(x @ W2^T + b2)   (M=2304, N=2048, K=2048)
  gemm_nt_wmma<<<dim3(V_DIMC / 64, (unsigned)(M / 64)), blk, 0, stream>>>(
      xb, V_DIMC, W2b, V_DIMC, b2, out, V_DIMC, V_DIMC, 1, 1);
}
